// SwitchMoELayer_26671746908368
// MI455X (gfx1250) — compile-verified, matrix operand order checked by reading, not run on hardware
//
#include <hip/hip_runtime.h>
#include <hip/hip_bf16.h>

// ---------------------------------------------------------------------------
// Switch-MoE (top-2 of 8 experts), routed implementation for gfx1250 (CDNA5).
// B=4, T=2048 -> N=8192 tokens, D=768, H=3072, E=8.
// Compute: f16 WMMA (v_wmma_f32_16x16x32_f16), f32 accumulate.
// GEMM tiling: 256 thr (8 waves), tile 64Mx64N, K-step 64, 2 accum/wave,
// 32 wmma (4/wave) per barrier. Fast path pre-transposes weights to f16 [E][N][K]
// once per launch (fits the 192MB L2) + double-buffered LDS staging.
// Staging helpers batch all global loads before converts/stores so the
// scheduler can pipeline s_wait_loadcnt.
// ---------------------------------------------------------------------------

#define NTOK 8192
#define DDIM 768
#define HDIM 3072
#define NEXP 8
// compacted assignment list: 2N real entries + per-expert padding to a
// 64-row tile boundary: 2*8192 + 8*63 = 16888 -> round up to 16896.
#define CAP  16896

typedef __attribute__((ext_vector_type(16))) _Float16 v16h;
typedef __attribute__((ext_vector_type(8)))  float    v8f;

// ---------------------------------------------------------------------------
// small kernels
// ---------------------------------------------------------------------------
__global__ void moe_init_kernel(int* __restrict__ counters /*16 ints*/,
                                int* __restrict__ tokl) {
    int i = blockIdx.x * 256 + threadIdx.x;
    if (i < 16)  counters[i] = 0;
    if (i < CAP) tokl[i] = -1;
}

__global__ void moe_zero_out_kernel(float4* __restrict__ out, int n4) {
    int i = blockIdx.x * 256 + threadIdx.x;
    if (i < n4) out[i] = make_float4(0.f, 0.f, 0.f, 0.f);
}

// gate: one wave per token -> logits, softmax, top-2
__global__ __launch_bounds__(256) void moe_gate_kernel(
    const float* __restrict__ x, const float* __restrict__ Wg,
    const float* __restrict__ bg, int* __restrict__ counts,
    int* __restrict__ top2idx, float* __restrict__ top2val) {
    const int wv   = threadIdx.x >> 5;
    const int lane = threadIdx.x & 31;
    const int n    = blockIdx.x * 8 + wv;
    if (n >= NTOK) return;

    const float* xr = x + (size_t)n * DDIM;
    float acc[NEXP];
#pragma unroll
    for (int e = 0; e < NEXP; ++e) acc[e] = 0.f;

    for (int d = lane; d < DDIM; d += 32) {
        float xv = xr[d];
#pragma unroll
        for (int e = 0; e < NEXP; ++e) acc[e] += xv * Wg[e * DDIM + d];
    }
#pragma unroll
    for (int e = 0; e < NEXP; ++e)
        for (int off = 16; off > 0; off >>= 1)
            acc[e] += __shfl_xor(acc[e], off, 32);

    if (lane == 0) {
        float l[NEXP], p[NEXP];
        float mx = -1e30f;
#pragma unroll
        for (int e = 0; e < NEXP; ++e) { l[e] = acc[e] + bg[e]; mx = fmaxf(mx, l[e]); }
        float s = 0.f;
#pragma unroll
        for (int e = 0; e < NEXP; ++e) { p[e] = __expf(l[e] - mx); s += p[e]; }
        float inv = 1.f / s;
#pragma unroll
        for (int e = 0; e < NEXP; ++e) p[e] *= inv;

        int i1 = 0; float p1 = p[0];
#pragma unroll
        for (int e = 1; e < NEXP; ++e) if (p[e] > p1) { p1 = p[e]; i1 = e; }
        int i2 = -1; float p2 = -1.f;
#pragma unroll
        for (int e = 0; e < NEXP; ++e) if (e != i1 && p[e] > p2) { p2 = p[e]; i2 = e; }

        top2idx[2 * n]     = i1;  top2val[2 * n]     = p1;
        top2idx[2 * n + 1] = i2;  top2val[2 * n + 1] = p2;
        atomicAdd(&counts[i1], 1);
        atomicAdd(&counts[i2], 1);
    }
}

__global__ void moe_scan_kernel(const int* __restrict__ counts,
                                int* __restrict__ offarr) {
    if (threadIdx.x == 0 && blockIdx.x == 0) {
        int o = 0;
        for (int e = 0; e < NEXP; ++e) {
            offarr[e] = o;
            o += (counts[e] + 63) & ~63;
        }
        offarr[NEXP] = o;
    }
}

__global__ void moe_scatter_kernel(const int* __restrict__ offarr,
                                   int* __restrict__ cursors,
                                   const int* __restrict__ top2idx,
                                   const float* __restrict__ top2val,
                                   int* __restrict__ tokl,
                                   float* __restrict__ wl) {
    int n = blockIdx.x * 256 + threadIdx.x;
    if (n >= NTOK) return;
#pragma unroll
    for (int j = 0; j < 2; ++j) {
        int e   = top2idx[2 * n + j];
        float w = top2val[2 * n + j];
        int s   = atomicAdd(&cursors[e], 1);
        int pos = offarr[e] + s;
        tokl[pos] = n;
        wl[pos]   = w;
    }
}

// weight transpose+convert: src f32 [E][K][N] -> dst f16 [E][N][K]
__global__ __launch_bounds__(256) void moe_transpose_kernel(
    const float* __restrict__ src, _Float16* __restrict__ dst,
    int K, int N) {
    __shared__ float tile[32][33];
    const int e  = blockIdx.z;
    const float*  S = src + (size_t)e * K * N;
    _Float16*     Dm = dst + (size_t)e * K * N;
    const int n0 = blockIdx.x * 32, k0 = blockIdx.y * 32;
    const int tx = threadIdx.x & 31, ty = threadIdx.x >> 5;   // ty: 0..7
#pragma unroll
    for (int j = 0; j < 4; ++j)
        tile[ty + 8 * j][tx] = S[(size_t)(k0 + ty + 8 * j) * N + n0 + tx];
    __syncthreads();
#pragma unroll
    for (int j = 0; j < 4; ++j)
        Dm[(size_t)(n0 + ty + 8 * j) * K + k0 + tx] = (_Float16)tile[tx][ty + 8 * j];
}

// ---------------------------------------------------------------------------
// Fragment helpers (ISA 7.12.2 wave32 layouts). A 16x32 f16: lane L holds
// row M=L%16; K pattern split by lane group (grp=L/16). B staged transposed
// (ldsB[n][k]) so its per-lane layout mirrors A. C/D: c[v] @ row grp*8+v,
// col L%16.
// ---------------------------------------------------------------------------
#define FRAG_KK(vi, grp) (((vi) < 4) ? ((grp) * 8 + (vi) * 2) \
                                     : (16 + (grp) * 8 + ((vi) - 4) * 2))

union HPack4 { _Float16 h[4]; uint2 u; };

__device__ __forceinline__ void build_frags(
    const _Float16 (*__restrict__ A)[72], const _Float16 (*__restrict__ B)[72],
    int wr, int wc0, int wc1, int grp, int l15,
    v16h& a0, v16h& a1, v16h& b00, v16h& b01, v16h& b10, v16h& b11) {
#pragma unroll
    for (int vi = 0; vi < 8; ++vi) {
        int kk = FRAG_KK(vi, grp);
        a0[2 * vi]      = A[wr + l15][kk];
        a0[2 * vi + 1]  = A[wr + l15][kk + 1];
        a1[2 * vi]      = A[wr + l15][kk + 32];
        a1[2 * vi + 1]  = A[wr + l15][kk + 33];
        b00[2 * vi]     = B[wc0 + l15][kk];
        b00[2 * vi + 1] = B[wc0 + l15][kk + 1];
        b01[2 * vi]     = B[wc0 + l15][kk + 32];
        b01[2 * vi + 1] = B[wc0 + l15][kk + 33];
        b10[2 * vi]     = B[wc1 + l15][kk];
        b10[2 * vi + 1] = B[wc1 + l15][kk + 1];
        b11[2 * vi]     = B[wc1 + l15][kk + 32];
        b11[2 * vi + 1] = B[wc1 + l15][kk + 33];
    }
}

__device__ __forceinline__ void wmma4(const v16h& a0, const v16h& a1,
                                      const v16h& b00, const v16h& b01,
                                      const v16h& b10, const v16h& b11,
                                      v8f& c0, v8f& c1) {
    c0 = __builtin_amdgcn_wmma_f32_16x16x32_f16(false, a0, false, b00,
                                                (short)0, c0, false, false);
    c1 = __builtin_amdgcn_wmma_f32_16x16x32_f16(false, a0, false, b10,
                                                (short)0, c1, false, false);
    c0 = __builtin_amdgcn_wmma_f32_16x16x32_f16(false, a1, false, b01,
                                                (short)0, c0, false, false);
    c1 = __builtin_amdgcn_wmma_f32_16x16x32_f16(false, a1, false, b11,
                                                (short)0, c1, false, false);
}

// staging helpers: batch ALL global loads first, then convert/store, so the
// compiler issues the loads back-to-back and drains with staged waits.
__device__ __forceinline__ void stage_A_gather(
    const float* const xrow[4], const float amsk[4], int k0,
    _Float16 (*__restrict__ dstA)[72], int ar, int f4) {
    float4 tmp[4];
#pragma unroll
    for (int j = 0; j < 4; ++j)
        tmp[j] = *reinterpret_cast<const float4*>(xrow[j] + k0 + f4 * 4);
#pragma unroll
    for (int j = 0; j < 4; ++j) {
        HPack4 pck;
        pck.h[0] = (_Float16)(tmp[j].x * amsk[j]);
        pck.h[1] = (_Float16)(tmp[j].y * amsk[j]);
        pck.h[2] = (_Float16)(tmp[j].z * amsk[j]);
        pck.h[3] = (_Float16)(tmp[j].w * amsk[j]);
        *reinterpret_cast<uint2*>(&dstA[ar + 16 * j][f4 * 4]) = pck.u;
    }
}

__device__ __forceinline__ void stage_h16(
    const _Float16* __restrict__ base /* row0 of 64-row panel */,
    size_t rowStride, int k0, _Float16 (*__restrict__ dst)[72],
    int br, int seg) {
    uint4 tmp[2];
#pragma unroll
    for (int j = 0; j < 2; ++j)
        tmp[j] = *reinterpret_cast<const uint4*>(
            base + (size_t)(br + 32 * j) * rowStride + k0 + seg * 8);
#pragma unroll
    for (int j = 0; j < 2; ++j)
        *reinterpret_cast<uint4*>(&dst[br + 32 * j][seg * 8]) = tmp[j];
}

__device__ __forceinline__ void stage_B_f32(
    const float* __restrict__ Wb /* +n0+nn4 */, size_t rowStride, int k0,
    _Float16 (*__restrict__ dstB)[72], int nn4, int kb) {
    float4 tmp[4];
#pragma unroll
    for (int j = 0; j < 4; ++j)
        tmp[j] = *reinterpret_cast<const float4*>(
            Wb + (size_t)(k0 + kb + 16 * j) * rowStride);
#pragma unroll
    for (int j = 0; j < 4; ++j) {
        int kk = kb + 16 * j;
        dstB[nn4 + 0][kk] = (_Float16)tmp[j].x;
        dstB[nn4 + 1][kk] = (_Float16)tmp[j].y;
        dstB[nn4 + 2][kk] = (_Float16)tmp[j].z;
        dstB[nn4 + 3][kk] = (_Float16)tmp[j].w;
    }
}

// ---------------------------------------------------------------------------
// FAST PATH GEMM1: Hact = GELU(gather(x) @ W1 + b1); W1T is f16 [E][H][D].
// Double-buffered LDS staging.
// ---------------------------------------------------------------------------
__global__ __launch_bounds__(256) void moe_gemm1t_kernel(
    const float* __restrict__ x, const _Float16* __restrict__ W1T,
    const float* __restrict__ b1, const int* __restrict__ offarr,
    const int* __restrict__ tokl, _Float16* __restrict__ Hact) {
    __shared__ __align__(16) _Float16 ldsA[2][64][72];
    __shared__ __align__(16) _Float16 ldsB[2][64][72];
    __shared__ int ldsTok[64];

    const int aBase = blockIdx.x * 64;
    if (aBase >= offarr[NEXP]) return;           // uniform exit: EXEC full
    int e = 0;
    while (aBase >= offarr[e + 1]) ++e;

    const int n0 = blockIdx.y * 64;
    const int t  = threadIdx.x;
    if (t < 64) ldsTok[t] = tokl[aBase + t];
    __syncthreads();

    const int lane = t & 31;
    const int w    = t >> 5;
    const int wr   = (w & 3) << 4;
    const int wc0  = (w >> 2) << 5;
    const int wc1  = wc0 + 16;
    const int grp  = lane >> 4;
    const int l15  = lane & 15;

    const int f4  = t & 15;                      // A gather geometry
    const int ar  = t >> 4;
    const int seg = t & 7;                       // B copy geometry
    const int br  = t >> 3;

    const float* xrow[4];
    float        amsk[4];
#pragma unroll
    for (int j = 0; j < 4; ++j) {
        int tok = ldsTok[ar + 16 * j];
        amsk[j] = (tok >= 0) ? 1.f : 0.f;
        xrow[j] = x + (size_t)(tok >= 0 ? tok : 0) * DDIM;
    }
    const _Float16* WbT = W1T + (size_t)e * DDIM * HDIM + (size_t)n0 * DDIM;

    v8f c0 = {}, c1 = {};
    stage_A_gather(xrow, amsk, 0, ldsA[0], ar, f4);
    stage_h16(WbT, DDIM, 0, ldsB[0], br, seg);

    int p = 0;
    for (int k0 = 0; k0 < DDIM; k0 += 64) {
        __syncthreads();
        if (k0 + 64 < DDIM) {
            stage_A_gather(xrow, amsk, k0 + 64, ldsA[p ^ 1], ar, f4);
            stage_h16(WbT, DDIM, k0 + 64, ldsB[p ^ 1], br, seg);
        }
        v16h a0, a1, b00, b01, b10, b11;
        build_frags(ldsA[p], ldsB[p], wr, wc0, wc1, grp, l15,
                    a0, a1, b00, b01, b10, b11);
        wmma4(a0, a1, b00, b01, b10, b11, c0, c1);
        p ^= 1;
    }

#pragma unroll
    for (int v = 0; v < 8; ++v) {
        int m  = wr + grp * 8 + v;
        int na = n0 + wc0 + l15;
        int nb = n0 + wc1 + l15;
        float z0 = c0[v] + b1[e * HDIM + na];
        float z1 = c1[v] + b1[e * HDIM + nb];
        float g0 = 0.5f * z0 * (1.0f + erff(z0 * 0.70710678118654752f));
        float g1 = 0.5f * z1 * (1.0f + erff(z1 * 0.70710678118654752f));
        Hact[(size_t)(aBase + m) * HDIM + na] = (_Float16)g0;
        Hact[(size_t)(aBase + m) * HDIM + nb] = (_Float16)g1;
    }
}

// ---------------------------------------------------------------------------
// FAST PATH GEMM2: out[tok] += w * (Hact @ W2 + b2); W2T is f16 [E][D][H].
// ---------------------------------------------------------------------------
__global__ __launch_bounds__(256) void moe_gemm2t_kernel(
    const _Float16* __restrict__ Hact, const _Float16* __restrict__ W2T,
    const float* __restrict__ b2, const int* __restrict__ offarr,
    const int* __restrict__ tokl, const float* __restrict__ wl,
    float* __restrict__ out) {
    __shared__ __align__(16) _Float16 ldsA[2][64][72];
    __shared__ __align__(16) _Float16 ldsB[2][64][72];
    __shared__ int   ldsTok[64];
    __shared__ float ldsW[64];

    const int aBase = blockIdx.x * 64;
    if (aBase >= offarr[NEXP]) return;
    int e = 0;
    while (aBase >= offarr[e + 1]) ++e;

    const int n0 = blockIdx.y * 64;
    const int t  = threadIdx.x;
    if (t < 64) { ldsTok[t] = tokl[aBase + t]; ldsW[t] = wl[aBase + t]; }
    __syncthreads();

    const int lane = t & 31;
    const int w    = t >> 5;
    const int wr   = (w & 3) << 4;
    const int wc0  = (w >> 2) << 5;
    const int wc1  = wc0 + 16;
    const int grp  = lane >> 4;
    const int l15  = lane & 15;

    const int seg = t & 7;
    const int br  = t >> 3;

    const _Float16* Arow0 = Hact + (size_t)aBase * HDIM;
    const _Float16* WbT   = W2T + (size_t)e * HDIM * DDIM + (size_t)n0 * HDIM;

    v8f c0 = {}, c1 = {};
    stage_h16(Arow0, HDIM, 0, ldsA[0], br, seg);
    stage_h16(WbT,   HDIM, 0, ldsB[0], br, seg);

    int p = 0;
    for (int k0 = 0; k0 < HDIM; k0 += 64) {
        __syncthreads();
        if (k0 + 64 < HDIM) {
            stage_h16(Arow0, HDIM, k0 + 64, ldsA[p ^ 1], br, seg);
            stage_h16(WbT,   HDIM, k0 + 64, ldsB[p ^ 1], br, seg);
        }
        v16h a0, a1, b00, b01, b10, b11;
        build_frags(ldsA[p], ldsB[p], wr, wc0, wc1, grp, l15,
                    a0, a1, b00, b01, b10, b11);
        wmma4(a0, a1, b00, b01, b10, b11, c0, c1);
        p ^= 1;
    }

#pragma unroll
    for (int v = 0; v < 8; ++v) {
        int m   = wr + grp * 8 + v;
        int tok = ldsTok[m];
        if (tok >= 0) {
            int na = n0 + wc0 + l15;
            int nb = n0 + wc1 + l15;
            float va = ldsW[m] * (c0[v] + b2[e * DDIM + na]);
            float vb = ldsW[m] * (c1[v] + b2[e * DDIM + nb]);
            unsafeAtomicAdd(&out[(size_t)tok * DDIM + na], va);
            unsafeAtomicAdd(&out[(size_t)tok * DDIM + nb], vb);
        }
    }
}

// ---------------------------------------------------------------------------
// FALLBACK GEMM1/GEMM2 (no transposed-weight buffer; f32 weights staged
// directly). Same math, single-buffered LDS.
// ---------------------------------------------------------------------------
__global__ __launch_bounds__(256) void moe_gemm1_kernel(
    const float* __restrict__ x, const float* __restrict__ W1,
    const float* __restrict__ b1, const int* __restrict__ offarr,
    const int* __restrict__ tokl, _Float16* __restrict__ Hact) {
    __shared__ __align__(16) _Float16 ldsA[64][72];
    __shared__ __align__(16) _Float16 ldsB[64][72];
    __shared__ int ldsTok[64];

    const int aBase = blockIdx.x * 64;
    if (aBase >= offarr[NEXP]) return;
    int e = 0;
    while (aBase >= offarr[e + 1]) ++e;

    const int n0 = blockIdx.y * 64;
    const int t  = threadIdx.x;
    if (t < 64) ldsTok[t] = tokl[aBase + t];
    __syncthreads();

    const int lane = t & 31;
    const int w    = t >> 5;
    const int wr   = (w & 3) << 4;
    const int wc0  = (w >> 2) << 5;
    const int wc1  = wc0 + 16;
    const int grp  = lane >> 4;
    const int l15  = lane & 15;

    const int f4  = t & 15;
    const int ar  = t >> 4;
    const int nn4 = (t & 15) * 4;
    const int kb  = t >> 4;

    const float* xrow[4];
    float        amsk[4];
#pragma unroll
    for (int j = 0; j < 4; ++j) {
        int tok = ldsTok[ar + 16 * j];
        amsk[j] = (tok >= 0) ? 1.f : 0.f;
        xrow[j] = x + (size_t)(tok >= 0 ? tok : 0) * DDIM;
    }
    const float* Wb = W1 + (size_t)e * DDIM * HDIM + n0 + nn4;

    v8f c0 = {}, c1 = {};
    for (int k0 = 0; k0 < DDIM; k0 += 64) {
        stage_A_gather(xrow, amsk, k0, ldsA, ar, f4);
        stage_B_f32(Wb, HDIM, k0, ldsB, nn4, kb);
        __syncthreads();
        v16h a0, a1, b00, b01, b10, b11;
        build_frags(ldsA, ldsB, wr, wc0, wc1, grp, l15,
                    a0, a1, b00, b01, b10, b11);
        wmma4(a0, a1, b00, b01, b10, b11, c0, c1);
        __syncthreads();
    }

#pragma unroll
    for (int v = 0; v < 8; ++v) {
        int m  = wr + grp * 8 + v;
        int na = n0 + wc0 + l15;
        int nb = n0 + wc1 + l15;
        float z0 = c0[v] + b1[e * HDIM + na];
        float z1 = c1[v] + b1[e * HDIM + nb];
        float g0 = 0.5f * z0 * (1.0f + erff(z0 * 0.70710678118654752f));
        float g1 = 0.5f * z1 * (1.0f + erff(z1 * 0.70710678118654752f));
        Hact[(size_t)(aBase + m) * HDIM + na] = (_Float16)g0;
        Hact[(size_t)(aBase + m) * HDIM + nb] = (_Float16)g1;
    }
}

__global__ __launch_bounds__(256) void moe_gemm2_kernel(
    const _Float16* __restrict__ Hact, const float* __restrict__ W2,
    const float* __restrict__ b2, const int* __restrict__ offarr,
    const int* __restrict__ tokl, const float* __restrict__ wl,
    float* __restrict__ out) {
    __shared__ __align__(16) _Float16 ldsA[64][72];
    __shared__ __align__(16) _Float16 ldsB[64][72];
    __shared__ int   ldsTok[64];
    __shared__ float ldsW[64];

    const int aBase = blockIdx.x * 64;
    if (aBase >= offarr[NEXP]) return;
    int e = 0;
    while (aBase >= offarr[e + 1]) ++e;

    const int n0 = blockIdx.y * 64;
    const int t  = threadIdx.x;
    if (t < 64) { ldsTok[t] = tokl[aBase + t]; ldsW[t] = wl[aBase + t]; }
    __syncthreads();

    const int lane = t & 31;
    const int w    = t >> 5;
    const int wr   = (w & 3) << 4;
    const int wc0  = (w >> 2) << 5;
    const int wc1  = wc0 + 16;
    const int grp  = lane >> 4;
    const int l15  = lane & 15;

    const int seg = t & 7;
    const int br  = t >> 3;
    const int nn4 = (t & 15) * 4;
    const int kb  = t >> 4;

    const _Float16* Arow0 = Hact + (size_t)aBase * HDIM;
    const float*    Wb    = W2 + (size_t)e * HDIM * DDIM + n0 + nn4;

    v8f c0 = {}, c1 = {};
    for (int k0 = 0; k0 < HDIM; k0 += 64) {
        stage_h16(Arow0, HDIM, k0, ldsA, br, seg);
        stage_B_f32(Wb, DDIM, k0, ldsB, nn4, kb);
        __syncthreads();
        v16h a0, a1, b00, b01, b10, b11;
        build_frags(ldsA, ldsB, wr, wc0, wc1, grp, l15,
                    a0, a1, b00, b01, b10, b11);
        wmma4(a0, a1, b00, b01, b10, b11, c0, c1);
        __syncthreads();
    }

#pragma unroll
    for (int v = 0; v < 8; ++v) {
        int m   = wr + grp * 8 + v;
        int tok = ldsTok[m];
        if (tok >= 0) {
            int na = n0 + wc0 + l15;
            int nb = n0 + wc1 + l15;
            float va = ldsW[m] * (c0[v] + b2[e * DDIM + na]);
            float vb = ldsW[m] * (c1[v] + b2[e * DDIM + nb]);
            unsafeAtomicAdd(&out[(size_t)tok * DDIM + na], va);
            unsafeAtomicAdd(&out[(size_t)tok * DDIM + nb], vb);
        }
    }
}

// ---------------------------------------------------------------------------
// host-side launcher
// ---------------------------------------------------------------------------
extern "C" void kernel_launch(void* const* d_in, const int* in_sizes, int n_in,
                              void* d_out, int out_size, void* d_ws, size_t ws_size,
                              hipStream_t stream) {
    const float* x  = (const float*)d_in[0];
    const float* Wg = (const float*)d_in[1];
    const float* bg = (const float*)d_in[2];
    const float* W1 = (const float*)d_in[3];
    const float* b1 = (const float*)d_in[4];
    const float* W2 = (const float*)d_in[5];
    const float* b2 = (const float*)d_in[6];
    float* out = (float*)d_out;

    char* ws = (char*)d_ws;
    // layout (bytes):
    //   0         counts[8]+cursors[8]
    //   64        offarr[9]
    //   128       top2idx[2N]  (65536)
    //   65664     top2val[2N]  (65536)
    //   131200    tokl[CAP]    (67584)
    //   198784    wl[CAP]      (67584)
    //   266752    Hact[CAP*H]  f16  (103,809,024)
    //   104075776 W1T f16 [E][H][D] (37,748,736)   } fast path only
    //   141824512 W2T f16 [E][D][H] (37,748,736)   }
    int*      counters = (int*)(ws + 0);
    int*      counts   = counters;
    int*      cursors  = counters + 8;
    int*      offarr   = (int*)(ws + 64);
    int*      top2idx  = (int*)(ws + 128);
    float*    top2val  = (float*)(ws + 128 + 65536);
    int*      tokl     = (int*)(ws + 131200);
    float*    wl       = (float*)(ws + 198784);
    _Float16* Hact     = (_Float16*)(ws + 266752);
    _Float16* W1T      = (_Float16*)(ws + 104075776ULL);
    _Float16* W2T      = (_Float16*)(ws + 141824512ULL);
    const size_t need_full = 179573248ULL;

    moe_init_kernel<<<(CAP + 255) / 256, 256, 0, stream>>>(counters, tokl);
    moe_zero_out_kernel<<<(NTOK * DDIM / 4 + 255) / 256, 256, 0, stream>>>(
        (float4*)out, NTOK * DDIM / 4);
    moe_gate_kernel<<<NTOK / 8, 256, 0, stream>>>(x, Wg, bg, counts,
                                                  top2idx, top2val);
    moe_scan_kernel<<<1, 32, 0, stream>>>(counts, offarr);
    moe_scatter_kernel<<<NTOK / 256, 256, 0, stream>>>(offarr, cursors,
                                                       top2idx, top2val,
                                                       tokl, wl);
    if (ws_size >= need_full) {
        // pre-convert weights to f16 [E][N][K] (one pass, ~226 MB traffic)
        moe_transpose_kernel<<<dim3(HDIM / 32, DDIM / 32, NEXP), 256, 0, stream>>>(
            W1, W1T, DDIM, HDIM);
        moe_transpose_kernel<<<dim3(DDIM / 32, HDIM / 32, NEXP), 256, 0, stream>>>(
            W2, W2T, HDIM, DDIM);
        moe_gemm1t_kernel<<<dim3(CAP / 64, HDIM / 64), 256, 0, stream>>>(
            x, W1T, b1, offarr, tokl, Hact);
        moe_gemm2t_kernel<<<dim3(CAP / 64, DDIM / 64), 256, 0, stream>>>(
            Hact, W2T, b2, offarr, tokl, wl, out);
    } else {
        moe_gemm1_kernel<<<dim3(CAP / 64, HDIM / 64), 256, 0, stream>>>(
            x, W1, b1, offarr, tokl, Hact);
        moe_gemm2_kernel<<<dim3(CAP / 64, DDIM / 64), 256, 0, stream>>>(
            Hact, W2, b2, offarr, tokl, wl, out);
    }
}
